// Encoder_11879879542052
// MI455X (gfx1250) — matrix operand order repaired and checked by reading, not compiled, not run on hardware
//
#include <hip/hip_runtime.h>
#include <stdint.h>
#include <stddef.h>

#define B_  64
#define T_  512
#define E_  256
#define H_  256
#define H3_ 768
#define HPAD 264   // 256 + 8 bf16 pad -> 528B row stride, 4-bank rotation

typedef __attribute__((ext_vector_type(16))) __bf16       v16bf;
typedef __attribute__((ext_vector_type(4)))  __bf16       bf16x4;
typedef __attribute__((ext_vector_type(8)))  float        v8f;
typedef __attribute__((ext_vector_type(4)))  float        f32x4;
typedef __attribute__((ext_vector_type(4)))  unsigned int u32x4;

union BF16x16 {
  v16bf v;
  __bf16 s[16];
  u32x4 q[2];
};

__device__ __forceinline__ v8f wmma_bf16(v16bf a, v16bf b, v8f c) {
  return __builtin_amdgcn_wmma_f32_16x16x32_bf16(false, a, false, b, (short)0, c,
                                                 false, false);
}
__device__ __forceinline__ float sigmoid_(float v) {
  return 1.0f / (1.0f + __expf(-v));
}
__device__ __forceinline__ float tanh_(float v) {
  float e = __expf(-2.0f * v);
  return (1.0f - e) / (1.0f + e);
}

// ---------------------------------------------------------------------------
// Convert the f32 embedding table (32000 x 256) to bf16, 4 elements/thread.
// ---------------------------------------------------------------------------
__global__ __launch_bounds__(256) void embcvt_kernel(
    const float* __restrict__ e, __bf16* __restrict__ o) {
  size_t i = ((size_t)blockIdx.x * 256 + threadIdx.x) * 4;
  f32x4 f = *(const f32x4*)(e + i);
  bf16x4 b;
  b.x = (__bf16)f.x; b.y = (__bf16)f.y; b.z = (__bf16)f.z; b.w = (__bf16)f.w;
  *(bf16x4*)(o + i) = b;
}

// ---------------------------------------------------------------------------
// Convert a (256 x 768) f32 weight matrix into bf16 WMMA B-fragments.
// Fragment (nt, kb): 32x16 bf16 tile; lane l's 16 values contiguous:
//   element i of lane l:  K = kb*32 + (i>>3)*16 + (l>>4)*8 + (i&7),  N = nt*16 + (l&15)
// Fragments ordered nt-major so the scan's kb loop streams contiguously.
// ---------------------------------------------------------------------------
__global__ __launch_bounds__(256) void wfrag_kernel(
    const float* __restrict__ w, __bf16* __restrict__ frag) {
  int t  = blockIdx.x * 256 + threadIdx.x;   // 0 .. 196607 (exact grid)
  int i  = t & 15;
  int l  = (t >> 4) & 31;
  int kb = (t >> 9) & 7;
  int nt = t >> 12;
  int K  = kb * 32 + ((i >> 3) << 4) + ((l >> 4) << 3) + (i & 7);
  int N  = nt * 16 + (l & 15);
  frag[t] = (__bf16)w[(size_t)K * H3_ + N];
}

// ---------------------------------------------------------------------------
// Fused embedding-gather + input projection GEMM:
//   xp[ts][b][n] = sum_e emb_bf16[x[b,t], e] * kernel[e, n] + bias0[n]
// grid = (2048/8, 48 ntiles, 2 dirs), block = 256 (8 waves, 1 M-tile/wave).
// B-fragments for this nt (8 KB) staged into LDS with async-to-LDS copies.
// ---------------------------------------------------------------------------
__global__ __launch_bounds__(256) void xproj_kernel(
    const int* __restrict__ x, const __bf16* __restrict__ embbf,
    const __bf16* __restrict__ kff, const __bf16* __restrict__ kfb,
    const float* __restrict__ bias_f, const float* __restrict__ bias_b,
    float* __restrict__ xpf, float* __restrict__ xpb) {
  const int dir = blockIdx.z;
  const __bf16* kf = dir ? kfb : kff;
  const float* bias0 = dir ? bias_b : bias_f;
  float* xp = dir ? xpb : xpf;
  const int nt   = blockIdx.y;
  const int tid  = threadIdx.x;
  const int lane = tid & 31;
  const int w    = tid >> 5;
  const int hi   = lane >> 4;

  __shared__ alignas(64) __bf16 bsm[8 * 512];  // 8 KB: 8 kb-fragments

  {  // async stage: 2 x 16B per thread covers 8192 B
    const __bf16* src = kf + (size_t)nt * (8 * 512);
    unsigned voff = (unsigned)(tid * 16);
    unsigned ldsa = voff;  // bsm is the only LDS object -> offset 0 base
    asm volatile(
        "global_load_async_to_lds_b128 %0, %1, %2\n\t"
        "global_load_async_to_lds_b128 %0, %1, %2 offset:4096\n\t"
        "s_wait_asynccnt 0"
        :: "v"(ldsa), "v"(voff), "s"(src)
        : "memory");
  }
  __syncthreads();

  const int mtile = blockIdx.x * 8 + w;          // 0..2047
  const int mrow  = mtile * 16 + (lane & 15);    // flattened (b,t) row
  const int tok   = x[mrow];
  const __bf16* row = embbf + (size_t)tok * E_ + hi * 8;

  v8f acc = {};
#pragma unroll
  for (int kb = 0; kb < 8; ++kb) {
    const u32x4* ap = (const u32x4*)(row + kb * 32);  // const-imm offsets
    BF16x16 af;
    af.q[0] = ap[0];   // K = k0 .. k0+7
    af.q[1] = ap[2];   // K = k0+16 .. k0+23
    v16bf bf = *(const v16bf*)&bsm[kb * 512 + lane * 16];
    acc = wmma_bf16(af.v, bf, acc);
  }

  const float bv  = bias0[nt * 16 + (lane & 15)];
  const int bidx  = (mtile * 16) >> 9;           // batch (blocks never straddle)
  const int tbase = (mtile * 16) & 511;
  const int ts0   = dir ? (T_ - 1 - (tbase + hi * 8)) : (tbase + hi * 8);
  float* op = xp + ((size_t)ts0 * B_ + bidx) * H3_ + nt * 16 + (lane & 15);
  // Constant +/- immediate offsets (r * 196608 B) instead of a pointer chain.
  if (dir == 0) {
#pragma unroll
    for (int r = 0; r < 8; ++r) op[(ptrdiff_t)r * (B_ * H3_)] = acc[r] + bv;
  } else {
#pragma unroll
    for (int r = 0; r < 8; ++r) op[-(ptrdiff_t)r * (B_ * H3_)] = acc[r] + bv;
  }
}

// ---------------------------------------------------------------------------
// Persistent recurrent scan: grid = 2 workgroups (fwd/bwd), 1024 thr = 32 waves.
// h lives in LDS (bf16). Per step: each wave owns 2 (mt,jt) units; per unit 3
// gate tiles x 8 K-block WMMAs. Single h buffer + two barriers/step (results
// held in registers between them). All hot-loop loads/stores use one base
// pointer per object + constant immediate offsets.
// ---------------------------------------------------------------------------
__global__ __launch_bounds__(1024) void gru_scan_kernel(
    const float* __restrict__ hidden,
    const __bf16* __restrict__ rkf, const __bf16* __restrict__ rkb,
    const float* __restrict__ bias_f, const float* __restrict__ bias_b,
    const float* __restrict__ xpf, const float* __restrict__ xpb,
    float* __restrict__ out) {
  const int dir = blockIdx.x;
  const __bf16* rk = dir ? rkb : rkf;
  const float* bias1 = (dir ? bias_b : bias_f) + H3_;
  const float* xp    = dir ? xpb : xpf;

  __shared__ alignas(64) __bf16 hsm[B_ * HPAD];

  const int tid  = threadIdx.x;
  const int lane = tid & 31;
  const int w    = tid >> 5;
  const int hi   = lane >> 4;
  const int jcol = lane & 15;

  for (int i = tid; i < B_ * H_; i += 1024) {
    int b = i >> 8, j = i & 255;
    hsm[b * HPAD + j] = (__bf16)hidden[i];
  }
  __syncthreads();

  const int mt    = (2 * w) >> 4;     // 0..3
  const int jtA   = (2 * w) & 15;     // first of two consecutive j-tiles
  const int mbase = mt * 16 + hi * 8; // first output row (batch idx) this lane
  const int mrow  = mt * 16 + jcol;   // A-operand row this lane gathers

  float bz[2], br[2], bh[2];
#pragma unroll
  for (int u = 0; u < 2; ++u) {
    int j = (jtA + u) * 16 + jcol;
    bz[u] = bias1[j];
    br[u] = bias1[H_ + j];
    bh[u] = bias1[2 * H_ + j];
  }

  // Invariant base pointers; everything in the t-loop is base + const-imm.
  const __bf16* aBase = hsm + mrow * HPAD + hi * 8;           // A-frag gather
  const __bf16* fb    = rk + (size_t)jtA * 4096 + lane * 16;  // gate-z frags
  __bf16* hsB = hsm + mbase * HPAD + jtA * 16 + jcol;         // h read/write
  const float* xq = xp + (size_t)mbase * H3_ + jtA * 16 + jcol;
  float* yq = out + ((size_t)mbase * T_ + (dir ? (T_ - 1) : 0)) * (2 * H_) +
              dir * H_ + jtA * 16 + jcol;
  const ptrdiff_t ystep = dir ? -(ptrdiff_t)(2 * H_) : (ptrdiff_t)(2 * H_);

  float hnv[2][8];
  for (int t = 0; t < T_; ++t) {
    // ---- A fragments (h) from LDS, all 8 K-blocks, reused across 6 tiles
    v16bf a[8];
#pragma unroll
    for (int kb = 0; kb < 8; ++kb) {
      BF16x16 af;
      const u32x4* p = (const u32x4*)(aBase + kb * 32);
      af.q[0] = p[0];   // K = k0 .. k0+7
      af.q[1] = p[2];   // K = k0+16 .. k0+23
      a[kb] = af.v;
    }
    if (t + 1 < T_) {  // global_prefetch_b8 of next step's xp slab
      __builtin_prefetch(xq + (B_ * H3_), 0, 1);
    }

#pragma unroll
    for (int u = 0; u < 2; ++u) {
      const __bf16* f = fb + u * 4096;
      v8f az = {}, ar = {}, ah = {};
#pragma unroll
      for (int kb = 0; kb < 8; ++kb) {
        az = wmma_bf16(a[kb], *(const v16bf*)(f + kb * 512), az);
        ar = wmma_bf16(a[kb], *(const v16bf*)(f + 65536 + kb * 512), ar);
        ah = wmma_bf16(a[kb], *(const v16bf*)(f + 131072 + kb * 512), ah);
      }
      const float* xu = xq + u * 16;
#pragma unroll
      for (int r = 0; r < 8; ++r) {
        float xz = xu[r * H3_];
        float xr = xu[r * H3_ + H_];
        float xh = xu[r * H3_ + 2 * H_];
        float z     = sigmoid_(xz + az[r] + bz[u]);
        float rg    = sigmoid_(xr + ar[r] + br[u]);
        float hcand = tanh_(xh + rg * (ah[r] + bh[u]));
        float hprev = (float)hsB[u * 16 + r * HPAD];
        float hn    = z * hprev + (1.0f - z) * hcand;
        hnv[u][r]   = hn;
        yq[u * 16 + (size_t)r * (T_ * 2 * H_)] = hn;  // 1MB const-imm strides
      }
    }
    __syncthreads();  // all reads of h done
#pragma unroll
    for (int u = 0; u < 2; ++u)
#pragma unroll
      for (int r = 0; r < 8; ++r)
        hsB[u * 16 + r * HPAD] = (__bf16)hnv[u][r];
    __syncthreads();  // h fully updated for next step

    xq += B_ * H3_;
    yq += ystep;
  }

  // Final hidden-state outputs (h_f, h_b): last step's hnv survives the loop.
  float* hout = out + (size_t)B_ * T_ * (2 * H_) + (size_t)dir * B_ * H_ +
                (size_t)mbase * H_ + jtA * 16 + jcol;
#pragma unroll
  for (int u = 0; u < 2; ++u)
#pragma unroll
    for (int r = 0; r < 8; ++r)
      hout[u * 16 + r * H_] = hnv[u][r];
}

// ---------------------------------------------------------------------------
extern "C" void kernel_launch(void* const* d_in, const int* in_sizes, int n_in,
                              void* d_out, int out_size, void* d_ws, size_t ws_size,
                              hipStream_t stream) {
  (void)in_sizes; (void)n_in; (void)out_size; (void)ws_size;
  const int*   x        = (const int*)  d_in[0];
  const float* hidden   = (const float*)d_in[1];
  const float* emb      = (const float*)d_in[2];
  const float* kernel_f = (const float*)d_in[3];
  const float* rkern_f  = (const float*)d_in[4];
  const float* bias_f   = (const float*)d_in[5];
  const float* kernel_b = (const float*)d_in[6];
  const float* rkern_b  = (const float*)d_in[7];
  const float* bias_b   = (const float*)d_in[8];
  float* out = (float*)d_out;

  char* ws = (char*)d_ws;
  const size_t FRAG_ELEMS = (size_t)8 * 48 * 32 * 16;  // 196608 bf16 per matrix
  __bf16* kff = (__bf16*)ws;
  __bf16* kfb = kff + FRAG_ELEMS;
  __bf16* rkf = kfb + FRAG_ELEMS;
  __bf16* rkb = rkf + FRAG_ELEMS;
  __bf16* embbf = rkb + FRAG_ELEMS;                    // 32000*256 bf16
  float* xpf = (float*)((char*)(embbf + (size_t)32000 * E_));
  float* xpb = xpf + (size_t)T_ * B_ * H3_;

  embcvt_kernel<<<(32000 * E_) / 1024, 256, 0, stream>>>(emb, embbf);
  wfrag_kernel<<<768, 256, 0, stream>>>(kernel_f, kff);
  wfrag_kernel<<<768, 256, 0, stream>>>(kernel_b, kfb);
  wfrag_kernel<<<768, 256, 0, stream>>>(rkern_f, rkf);
  wfrag_kernel<<<768, 256, 0, stream>>>(rkern_b, rkb);
  xproj_kernel<<<dim3(256, 48, 2), 256, 0, stream>>>(
      x, embbf, kff, kfb, bias_f, bias_b, xpf, xpb);
  gru_scan_kernel<<<2, 1024, 0, stream>>>(
      hidden, rkf, rkb, bias_f, bias_b, xpf, xpb, out);
}
